// AtomAttentionEncoder_33921651704330
// MI455X (gfx1250) — compile-verified, matrix operand order checked by reading, not run on hardware
//
#include <hip/hip_runtime.h>
#include <hip/hip_bf16.h>

// ---------------------------------------------------------------------------
// Types for CDNA5 WMMA
// ---------------------------------------------------------------------------
typedef __attribute__((ext_vector_type(16))) __bf16 bf16x16;
typedef __attribute__((ext_vector_type(8)))  float  f32x8;
typedef __attribute__((ext_vector_type(4)))  unsigned int u32x4;

#define N_ATOMS 8192
#define N_TOK   1024
#define C_ATOM  128
#define C_OUT   384
#define N_HEADS 4
#define HEAD_D  32

// q pre-scale: (1/sqrt(32)) * log2(e)  -> lets us use exp2 (v_exp_f32) directly
#define QSCALE (0.17677669529663687f * 1.4426950408889634f)

__device__ __forceinline__ unsigned short f2bf(float f) {
    unsigned int u = __float_as_uint(f);
    unsigned int r = u + 0x7FFFu + ((u >> 16) & 1u);   // round-to-nearest-even
    return (unsigned short)(r >> 16);
}

// Load a 16x32 (rows x K) operand tile in the CDNA5 16-bit A/B VGPR layout.
// lane (half,l):  elements 0..7  = K = kb + half*8 + e
//                 elements 8..15 = K = kb + 16 + half*8 + (e-8)
// `base` is bf16 (stored as ushort), row-major with `stride` elements per row.
__device__ __forceinline__ bf16x16 load_tile(const unsigned short* base,
                                             int row, int stride, int kb, int half) {
    union { bf16x16 v; u32x4 q[2]; } u;
    const unsigned short* p = base + (size_t)row * stride + kb + half * 8;
    u.q[0] = *(const u32x4*)(p);
    u.q[1] = *(const u32x4*)(p + 16);
    return u.v;
}

__device__ __forceinline__ f32x8 wmma_bf16(bf16x16 a, bf16x16 b, f32x8 c) {
    return __builtin_amdgcn_wmma_f32_16x16x32_bf16(false, a, false, b,
                                                   (short)0, c, false, false);
}

// ---------------------------------------------------------------------------
// 1) Embed: h = [pos|elem] @ W_proj + b_proj   (fp32 + bf16 copies)
// ---------------------------------------------------------------------------
__global__ void embed_kernel(const float* __restrict__ pos,
                             const float* __restrict__ elem,
                             const float* __restrict__ Wp,
                             const float* __restrict__ bp,
                             float* __restrict__ h,
                             unsigned short* __restrict__ hb) {
    int n = blockIdx.x, c = threadIdx.x;
    float acc = bp[c];
    acc += pos[n * 3 + 0] * Wp[0 * C_ATOM + c];
    acc += pos[n * 3 + 1] * Wp[1 * C_ATOM + c];
    acc += pos[n * 3 + 2] * Wp[2 * C_ATOM + c];
    const float* er = elem + (size_t)n * 128;
#pragma unroll 8
    for (int k = 0; k < 128; ++k) acc += er[k] * Wp[(k + 3) * C_ATOM + c];
    h[(size_t)n * C_ATOM + c] = acc;
    hb[(size_t)n * C_ATOM + c] = f2bf(acc);
}

// ---------------------------------------------------------------------------
// 2) Transpose + bf16-convert the four 128x128 weights (Wq,Wk,Wv,Wo)
//    Wt[w][n*128+k] = bf16(W[w][k*128+n])  -> B-operand rows are contiguous
// ---------------------------------------------------------------------------
__global__ void prepw_kernel(const float* __restrict__ Wq, const float* __restrict__ Wk,
                             const float* __restrict__ Wv, const float* __restrict__ Wo,
                             unsigned short* __restrict__ Wt) {
    int id = blockIdx.x * blockDim.x + threadIdx.x;      // 65536 total
    int w = id >> 14, r = id & 16383, k = r >> 7, n = r & 127;
    const float* W = (w == 0) ? Wq : (w == 1) ? Wk : (w == 2) ? Wv : Wo;
    Wt[w * 16384 + n * 128 + k] = f2bf(W[k * 128 + n]);
}

__global__ void zero_kernel(float* __restrict__ p, int n) {
    int i = blockIdx.x * blockDim.x + threadIdx.x;
    if (i < n) p[i] = 0.0f;
}

// ---------------------------------------------------------------------------
// 3) WMMA projection: Y = A(bf16 [8192,128]) @ Wt^T + bias
//    mode 0: q  -> bf16 row-major, scaled by QSCALE
//    mode 1: k  -> bf16 row-major
//    mode 2: v  -> bf16 TRANSPOSED  vT[dim][atom]  (dim = head*32+d)
//    mode 3: x  -> f32 row-major, += residual h
//    One wave per 16x16 output tile; K-loop 128/32 = 4 WMMAs.
// ---------------------------------------------------------------------------
__global__ void proj_kernel(const unsigned short* __restrict__ A,
                            const unsigned short* __restrict__ Bt,
                            const float* __restrict__ bias,
                            const float* __restrict__ hres,
                            float* __restrict__ outf,
                            unsigned short* __restrict__ outb,
                            int mode) {
    const int lane = threadIdx.x & 31;
    const int half = lane >> 4, l = lane & 15;
    const int gw = blockIdx.x * 8 + (threadIdx.x >> 5);  // 4096 waves
    const int m0 = (gw >> 3) << 4;                       // 512 M tiles
    const int n0 = (gw & 7) << 4;                        // 8  N tiles

    f32x8 c = {};
#pragma unroll
    for (int kb = 0; kb < 128; kb += 32) {
        bf16x16 a = load_tile(A,  m0 + l, 128, kb, half);
        bf16x16 b = load_tile(Bt, n0 + l, 128, kb, half);
        c = wmma_bf16(a, b, c);
    }
    const float bv = bias[n0 + l];
    // C layout: vgpr j, lane(half,l) -> row m0+j+8*half, col n0+l
    if (mode == 0) {
#pragma unroll
        for (int j = 0; j < 8; ++j)
            outb[(size_t)(m0 + j + 8 * half) * 128 + n0 + l] = f2bf((c[j] + bv) * QSCALE);
    } else if (mode == 1) {
#pragma unroll
        for (int j = 0; j < 8; ++j)
            outb[(size_t)(m0 + j + 8 * half) * 128 + n0 + l] = f2bf(c[j] + bv);
    } else if (mode == 2) {
#pragma unroll
        for (int j = 0; j < 8; ++j)
            outb[(size_t)(n0 + l) * N_ATOMS + m0 + j + 8 * half] = f2bf(c[j] + bv);
    } else {
#pragma unroll
        for (int j = 0; j < 8; ++j) {
            size_t idx = (size_t)(m0 + j + 8 * half) * 128 + n0 + l;
            outf[idx] = c[j] + bv + hres[idx];
        }
    }
}

// ---------------------------------------------------------------------------
// 4) Flash attention, transposed formulation.
//    Per wave: one (head, 16-query) tile, loop over 8192 keys in chunks of 32.
//    S^T = K_tile x Q^T  (A = K rows, B = Q rows)  -> scores per query IN-LANE
//    O^T += V^T_tile x P^T (P^T built in-lane from S^T C-layout: it IS B-layout)
// ---------------------------------------------------------------------------
__global__ void attn_kernel(const unsigned short* __restrict__ qm,
                            const unsigned short* __restrict__ km,
                            const unsigned short* __restrict__ vT,
                            unsigned short* __restrict__ attnb) {
    const int lane = threadIdx.x & 31;
    const int half = lane >> 4, l = lane & 15;
    const int gw = blockIdx.x * 8 + (threadIdx.x >> 5);  // 2048 waves
    const int head = gw >> 9;
    const int q0 = (gw & 511) << 4;
    const int hd = head * HEAD_D;

    // Q operand (B layout): loop-invariant, lane l holds query q0+l over 32 dims
    const bf16x16 qb = load_tile(qm, q0 + l, 128, hd, half);

    float m = -3.0e38f, lsum = 0.0f;
    f32x8 oc0 = {}, oc1 = {};

    for (int kc = 0; kc < N_ATOMS; kc += 32) {
        bf16x16 a0 = load_tile(km, kc + l,      128, hd, half);
        bf16x16 a1 = load_tile(km, kc + 16 + l, 128, hd, half);
        f32x8 z = {};
        f32x8 s0 = wmma_bf16(a0, qb, z);   // keys kc..kc+15   x queries
        f32x8 s1 = wmma_bf16(a1, qb, z);   // keys kc+16..kc+31 x queries

        // per-query (lane) max over this 32-key chunk: 15 in-lane + 1 shuffle
        float cm = s0[0];
#pragma unroll
        for (int j = 1; j < 8; ++j) cm = fmaxf(cm, s0[j]);
#pragma unroll
        for (int j = 0; j < 8; ++j) cm = fmaxf(cm, s1[j]);
        cm = fmaxf(cm, __shfl_xor(cm, 16, 32));

        float mn    = fmaxf(m, cm);
        float alpha = exp2f(m - mn);

        float p0[8], p1[8], cs = 0.0f;
#pragma unroll
        for (int j = 0; j < 8; ++j) { p0[j] = exp2f(s0[j] - mn); cs += p0[j]; }
#pragma unroll
        for (int j = 0; j < 8; ++j) { p1[j] = exp2f(s1[j] - mn); cs += p1[j]; }
        cs += __shfl_xor(cs, 16, 32);

        lsum = lsum * alpha + cs;
        m = mn;
#pragma unroll
        for (int j = 0; j < 8; ++j) { oc0[j] *= alpha; oc1[j] *= alpha; }

        // P^T B-operand: S^T C-layout already matches B-layout -> in-lane pack
        union { bf16x16 v; unsigned short s[16]; } pb;
#pragma unroll
        for (int j = 0; j < 8; ++j) { pb.s[j] = f2bf(p0[j]); pb.s[8 + j] = f2bf(p1[j]); }

        bf16x16 va0 = load_tile(vT, hd + l,      N_ATOMS, kc, half); // dims 0..15
        bf16x16 va1 = load_tile(vT, hd + 16 + l, N_ATOMS, kc, half); // dims 16..31
        oc0 = wmma_bf16(va0, pb.v, oc0);
        oc1 = wmma_bf16(va1, pb.v, oc1);
    }

    const float inv = 1.0f / lsum;
    // O^T C layout: vgpr j -> dim = dt*16 + j + 8*half, query = l
#pragma unroll
    for (int j = 0; j < 8; ++j) {
        attnb[(size_t)(q0 + l) * 128 + hd + j + 8 * half]      = f2bf(oc0[j] * inv);
        attnb[(size_t)(q0 + l) * 128 + hd + 16 + j + 8 * half] = f2bf(oc1[j] * inv);
    }
}

// ---------------------------------------------------------------------------
// 5) LayerNorm per atom row + atomic segment-sum into tokens
// ---------------------------------------------------------------------------
__global__ void ln_seg_kernel(const float* __restrict__ x,
                              const float* __restrict__ g, const float* __restrict__ b,
                              const int* __restrict__ a2t,
                              float* __restrict__ toksum, float* __restrict__ tokcnt) {
    int n = blockIdx.x, c = threadIdx.x;
    float v = x[(size_t)n * 128 + c];
    __shared__ float red[128];
    red[c] = v; __syncthreads();
    for (int off = 64; off > 0; off >>= 1) { if (c < off) red[c] += red[c + off]; __syncthreads(); }
    float mu = red[0] * (1.0f / 128.0f); __syncthreads();
    float d = v - mu;
    red[c] = d * d; __syncthreads();
    for (int off = 64; off > 0; off >>= 1) { if (c < off) red[c] += red[c + off]; __syncthreads(); }
    float var = red[0] * (1.0f / 128.0f);
    float y = d * rsqrtf(var + 1e-5f) * g[c] + b[c];
    int t = a2t[n];
    atomicAdd(&toksum[(size_t)t * 128 + c], y);
    if (c == 0) atomicAdd(&tokcnt[t], 1.0f);
}

// ---------------------------------------------------------------------------
// 6) Token output: out = (sums/max(cnt,1)) @ W_agg + b_agg   [1024,384]
// ---------------------------------------------------------------------------
__global__ void tok_kernel(const float* __restrict__ toksum, const float* __restrict__ tokcnt,
                           const float* __restrict__ Wagg, const float* __restrict__ bagg,
                           float* __restrict__ out) {
    int t = blockIdx.x, tid = threadIdx.x;   // 384 threads
    __shared__ float row[128];
    if (tid < 128) row[tid] = toksum[(size_t)t * 128 + tid] / fmaxf(tokcnt[t], 1.0f);
    __syncthreads();
    float acc = bagg[tid];
#pragma unroll 8
    for (int c = 0; c < 128; ++c) acc += row[c] * Wagg[c * C_OUT + tid];
    out[(size_t)t * C_OUT + tid] = acc;
}

// ---------------------------------------------------------------------------
extern "C" void kernel_launch(void* const* d_in, const int* in_sizes, int n_in,
                              void* d_out, int out_size, void* d_ws, size_t ws_size,
                              hipStream_t stream) {
    const float* ref_pos  = (const float*)d_in[0];
    const float* ref_elem = (const float*)d_in[1];
    const int*   a2t      = (const int*)  d_in[2];
    // d_in[3] = N_tokens scalar (1024, hardcoded)
    const float* W_proj = (const float*)d_in[4];
    const float* b_proj = (const float*)d_in[5];
    const float* Wq = (const float*)d_in[6];   const float* bq = (const float*)d_in[7];
    const float* Wk = (const float*)d_in[8];   const float* bk = (const float*)d_in[9];
    const float* Wv = (const float*)d_in[10];  const float* bv = (const float*)d_in[11];
    const float* Wo = (const float*)d_in[12];  const float* bo = (const float*)d_in[13];
    const float* ln_g = (const float*)d_in[14]; const float* ln_b = (const float*)d_in[15];
    const float* W_agg = (const float*)d_in[16]; const float* b_agg = (const float*)d_in[17];
    float* out = (float*)d_out;

    char* ws = (char*)d_ws;
    float*          h_f32  = (float*)         (ws + 0);          // 4 MB
    unsigned short* h_bf   = (unsigned short*)(ws + 4194304);    // 2 MB
    unsigned short* q_bf   = (unsigned short*)(ws + 6291456);    // 2 MB
    unsigned short* k_bf   = (unsigned short*)(ws + 8388608);    // 2 MB
    unsigned short* vT_bf  = (unsigned short*)(ws + 10485760);   // 2 MB [128][8192]
    unsigned short* at_bf  = (unsigned short*)(ws + 12582912);   // 2 MB attn out
    float*          x_f32  = (float*)         (ws + 14680064);   // 4 MB
    unsigned short* Wt     = (unsigned short*)(ws + 18874368);   // 4 x 32 KB (q,k,v,o)
    float*          toksum = (float*)         (ws + 19005440);   // 512 KB
    float*          tokcnt = (float*)         (ws + 19529728);   // 4 KB

    embed_kernel<<<N_ATOMS, 128, 0, stream>>>(ref_pos, ref_elem, W_proj, b_proj, h_f32, h_bf);
    prepw_kernel<<<256, 256, 0, stream>>>(Wq, Wk, Wv, Wo, Wt);
    zero_kernel<<<(N_TOK * 128 + N_TOK + 255) / 256, 256, 0, stream>>>(toksum, N_TOK * 128 + N_TOK);

    // q (scaled), k, v (transposed)  -- WMMA projections
    proj_kernel<<<512, 256, 0, stream>>>(h_bf, Wt + 0 * 16384, bq, nullptr, nullptr, q_bf, 0);
    proj_kernel<<<512, 256, 0, stream>>>(h_bf, Wt + 1 * 16384, bk, nullptr, nullptr, k_bf, 1);
    proj_kernel<<<512, 256, 0, stream>>>(h_bf, Wt + 2 * 16384, bv, nullptr, nullptr, vT_bf, 2);

    // flash attention (2048 waves = 256 blocks x 8 waves)
    attn_kernel<<<256, 256, 0, stream>>>(q_bf, k_bf, vT_bf, at_bf);

    // out-proj + residual  -> x
    proj_kernel<<<512, 256, 0, stream>>>(at_bf, Wt + 3 * 16384, bo, h_f32, x_f32, nullptr, 3);

    // LayerNorm + segment sum, then token GEMM
    ln_seg_kernel<<<N_ATOMS, 128, 0, stream>>>(x_f32, ln_g, ln_b, a2t, toksum, tokcnt);
    tok_kernel<<<N_TOK, C_OUT, 0, stream>>>(toksum, tokcnt, W_agg, b_agg, out);
}
